// mLSTMBlock_75926431859104
// MI455X (gfx1250) — matrix operand order repaired and checked by reading, not compiled, not run on hardware
//
#include <hip/hip_runtime.h>
#include <hip/hip_bf16.h>
#include <math.h>

typedef __attribute__((ext_vector_type(16))) __bf16 v16bf;
typedef __attribute__((ext_vector_type(8)))  float  v8f;
typedef __attribute__((ext_vector_type(4)))  float  f32x4;

#define Bdim 256
#define Ddim 512
#define Hdim 512

// Fast sigmoid: one v_exp_f32 + one v_rcp_f32 (avoid IEEE div expansion).
__device__ __forceinline__ float fast_sigmoid(float z) {
  return __builtin_amdgcn_rcpf(1.0f + __expf(-z));
}

// Fast tanh: tanh(x) = 1 - 2/(exp(2x)+1); exp via v_exp_f32, div via v_rcp_f32.
__device__ __forceinline__ float fast_tanh(float x) {
  float e = __expf(2.0f * x);
  return 1.0f - 2.0f * __builtin_amdgcn_rcpf(e + 1.0f);
}

// ---------------------------------------------------------------------------
// Kernel 1: fused 6-way projection GEMM  Y = x @ Wcat^T  (Wcat = [Wq;Wk;Wv;Wi;Wf;Wo])
// bf16 WMMA 16x16x32, f32 accumulate. Gates get bias + sigmoid in the epilogue.
// Block = 256 threads = 8 waves; each wave owns one 16x16 output tile.
// Grid: (B/16, 6H/128) = (16, 24).
// ---------------------------------------------------------------------------
__global__ __launch_bounds__(256) void mlstm_proj_kernel(
    const float* __restrict__ x,
    const float* __restrict__ Wq, const float* __restrict__ Wk,
    const float* __restrict__ Wv, const float* __restrict__ Wi,
    const float* __restrict__ Wf, const float* __restrict__ Wo,
    const float* __restrict__ bi, const float* __restrict__ bf,
    const float* __restrict__ bo,
    float* __restrict__ ws)             // 6 x [B,H] f32: q,k,v,is,fs,os
{
  const int lane = threadIdx.x & 31;
  const int wave = threadIdx.x >> 5;
  const int m0   = blockIdx.x * 16;               // batch tile
  const int n0   = blockIdx.y * 128 + wave * 16;  // column in concat (0..3071)
  const int seg  = n0 >> 9;                       // which of 6 weights (512|128)
  const int nl   = n0 & 511;                      // column within segment

  const float* W = (seg == 0) ? Wq : (seg == 1) ? Wk : (seg == 2) ? Wv
                 : (seg == 3) ? Wi : (seg == 4) ? Wf : Wo;

  const int half = lane >> 4;                  // 0: lanes 0-15, 1: lanes 16-31
  const int mr   = lane & 15;                  // A row for this lane
  const int nc   = lane & 15;                  // B column for this lane

  const float* arow = x + (size_t)(m0 + mr) * Ddim;
  const float* brow = W + (size_t)(nl + nc) * Ddim;

  v8f acc = {};
  for (int k0 = 0; k0 < Ddim; k0 += 32) {
    v16bf a, b;
    const float* ap = arow + k0 + half * 16;   // A: 16 consecutive K
    const float* bp = brow + k0 + half * 16;   // B: 16 consecutive K
#pragma unroll
    for (int e = 0; e < 8; ++e) {
      a[e]     = (__bf16)ap[e];
      a[e + 8] = (__bf16)ap[e + 8];
    }
#pragma unroll
    for (int e = 0; e < 16; ++e) b[e] = (__bf16)bp[e];
    acc = __builtin_amdgcn_wmma_f32_16x16x32_bf16(
        /*neg_a=*/false, a, /*neg_b=*/false, b,
        /*c_mod=*/(short)0, acc, /*reuse_a=*/false, /*reuse_b=*/false);
  }

  // Epilogue: D layout -> lane holds col (lane&15), rows r + 8*half.
  const int col = nl + nc;
  float bias = 0.f;
  if (seg >= 3) {
    const float* bb = (seg == 3) ? bi : (seg == 4) ? bf : bo;
    bias = bb[col];
  }
  float* outseg = ws + (size_t)seg * Bdim * Hdim;
#pragma unroll
  for (int r = 0; r < 8; ++r) {
    const int row = m0 + r + half * 8;
    float v = acc[r];
    if (seg >= 3) v = fast_sigmoid(v + bias);          // gate activation
    outseg[(size_t)row * Hdim + col] = v;
  }
}

// ---------------------------------------------------------------------------
// Kernel 2: streaming state update + fused readout. Memory-roofline bound:
// 256MB C_prev in + 256MB C_t out (~22us @ 23.3 TB/s). One wave per (b,i) row.
// Non-temporal C stream (bigger than 192MB L2, touched once); q/v rows stay
// hot in L1/L2. C_t never re-read: dot with q accumulated in-pass.
// ---------------------------------------------------------------------------
__global__ __launch_bounds__(256) void mlstm_state_kernel(
    const float* __restrict__ C_prev,
    const float* __restrict__ ws,
    float* __restrict__ h_out,          // [B,H]
    float* __restrict__ C_out)          // [B,H,H]
{
  const int lane   = threadIdx.x & 31;
  const int wave   = threadIdx.x >> 5;
  const int row_id = blockIdx.x * 8 + wave;     // 0 .. B*H-1
  const int b      = row_id >> 9;
  const int i      = row_id & 511;

  const size_t BH = (size_t)Bdim * Hdim;
  const float* qrow = ws            + (size_t)b * Hdim;
  const float* krow = ws + 1 * BH   + (size_t)b * Hdim;
  const float* vrow = ws + 2 * BH   + (size_t)b * Hdim;
  const float* irow = ws + 3 * BH   + (size_t)b * Hdim;
  const float* frow = ws + 4 * BH   + (size_t)b * Hdim;
  const float* orow = ws + 5 * BH   + (size_t)b * Hdim;

  const float fg = frow[i];
  const float ik = irow[i] * krow[i];

  const f32x4* Cp = (const f32x4*)(C_prev + (size_t)row_id * Hdim);
  f32x4*       Co = (f32x4*)(C_out + (size_t)row_id * Hdim);
  const f32x4* v4 = (const f32x4*)vrow;
  const f32x4* q4 = (const f32x4*)qrow;

  float acc = 0.f;
#pragma unroll
  for (int t = 0; t < 4; ++t) {
    const int idx = t * 32 + lane;              // contiguous 512B per b128 op
    f32x4 c  = __builtin_nontemporal_load(&Cp[idx]);
    f32x4 vv = v4[idx];
    f32x4 nc = fg * c + ik * vv;
    __builtin_nontemporal_store(nc, &Co[idx]);
    f32x4 p  = nc * q4[idx];
    acc += p.x + p.y + p.z + p.w;
  }
  // wave32 butterfly reduction
#pragma unroll
  for (int off = 16; off > 0; off >>= 1)
    acc += __shfl_xor(acc, off, 32);

  if (lane == 0)
    h_out[row_id] = orow[i] * fast_tanh(acc);
}

// ---------------------------------------------------------------------------
extern "C" void kernel_launch(void* const* d_in, const int* in_sizes, int n_in,
                              void* d_out, int out_size, void* d_ws, size_t ws_size,
                              hipStream_t stream) {
  const float* x      = (const float*)d_in[0];
  // d_in[1] = h_prev (unused by reference)
  const float* C_prev = (const float*)d_in[2];
  const float* Wq = (const float*)d_in[3];
  const float* Wk = (const float*)d_in[4];
  const float* Wv = (const float*)d_in[5];
  const float* Wi = (const float*)d_in[6];
  const float* Wf = (const float*)d_in[7];
  const float* Wo = (const float*)d_in[8];
  const float* bi = (const float*)d_in[9];
  const float* bf = (const float*)d_in[10];
  const float* bo = (const float*)d_in[11];

  float* out   = (float*)d_out;
  float* h_out = out;                                   // first B*H floats
  float* C_out = out + (size_t)Bdim * Hdim;             // then B*H*H
  float* ws    = (float*)d_ws;                          // 6*B*H f32 = 3MB

  dim3 g1(Bdim / 16, (6 * Hdim) / 128);                 // (16, 24)
  mlstm_proj_kernel<<<g1, 256, 0, stream>>>(x, Wq, Wk, Wv, Wi, Wf, Wo,
                                            bi, bf, bo, ws);

  const int rows = Bdim * Hdim;                         // 131072 waves
  mlstm_state_kernel<<<rows / 8, 256, 0, stream>>>(C_prev, ws, h_out, C_out);
}
